// EncoderLayer_35021163331623
// MI455X (gfx1250) — matrix operand order repaired and checked by reading, not compiled, not run on hardware
//
#include <hip/hip_runtime.h>
#include <hip/hip_bf16.h>

// ---------------- problem constants ----------------
constexpr int Bc   = 2;
constexpr int Sc   = 2048;
constexpr int Dc   = 512;
constexpr int Hc   = 8;
constexpr int DKc  = 128;
constexpr int DVc  = 256;
constexpr int DFFc = 2048;
constexpr int Mc   = Bc * Sc;            // 4096 rows
#define ATTN_SCALE 0.08838834764831845f  // 1/sqrt(128)

// ---------------- vector types ----------------
typedef __attribute__((ext_vector_type(16))) __bf16        v16bf;
typedef __attribute__((ext_vector_type(8)))  float         v8f;
typedef __attribute__((ext_vector_type(4)))  unsigned int  u32x4;
typedef __attribute__((ext_vector_type(4)))  float         f32x4;

union Frag {            // one WMMA 16-bit operand: 16 bf16 per lane = 8 VGPRs
    v16bf bf;
    u32x4 q[2];
};

__device__ __forceinline__ unsigned short f32_to_bf16(float f) {
    unsigned int u = __float_as_uint(f);
    u = (u + 0x7FFFu + ((u >> 16) & 1u)) >> 16;   // round-to-nearest-even
    return (unsigned short)u;
}

__device__ __forceinline__ v8f wmma_bf16(const Frag& a, const Frag& b, v8f c) {
    return __builtin_amdgcn_wmma_f32_16x16x32_bf16(
        false, a.bf, false, b.bf, (short)0, c, false, false);
}

// gfx1250 async DMA: 16B per lane, global (SGPR base + VGPR u32 offset) -> LDS.
// Tracked with ASYNCcnt; wait with s_wait_asynccnt.
__device__ __forceinline__ void async_copy_b128(unsigned lds_addr, unsigned goff,
                                                unsigned long long sbase) {
    asm volatile("global_load_async_to_lds_b128 %0, %1, %2"
                 :: "v"(lds_addr), "v"(goff), "s"(sbase) : "memory");
}
__device__ __forceinline__ void wait_async0() {
    asm volatile("s_wait_asynccnt 0x0" ::: "memory");
}

// =====================================================================
// Generic GEMM:  out[M,N] = A[M,K] @ W[K,N] (+bias)(+relu)(+residual)
//   HEAD_DIM>0 : W stacked per head [H, K, HEAD_DIM].
//   OUT_MODE: 0 = f32 [M,N]; 1 = bf16 [B,H,S,hd]; 2 = bf16 [B,H,hd,S].
// Tile 128x128x32, 256 threads = 8 waves (32x64 each), double-buffered LDS.
// =====================================================================
template<int HEAD_DIM, int OUT_MODE, bool RELU, bool RESID>
__global__ __launch_bounds__(256) void gemm_bf16_kernel(
    const float* __restrict__ A, const float* __restrict__ W,
    const float* __restrict__ bias, const float* __restrict__ resid,
    void* __restrict__ outp, int M, int N, int K, int seqS)
{
    constexpr int HD = (HEAD_DIM > 0) ? HEAD_DIM : 1;
    __shared__ __align__(16) unsigned short As[2][128][32];  // [m][k] bf16
    __shared__ __align__(16) unsigned short Bs[2][128][32];  // [n][k] bf16

    const int tid  = threadIdx.x;
    const int lane = tid & 31;
    const int wave = tid >> 5;
    const int m0   = blockIdx.y * 128;
    const int n0   = blockIdx.x * 128;
    const int wm   = (wave & 3) * 32;
    const int wn   = (wave >> 2) * 64;
    const int l16  = lane & 15;
    const int half = lane >> 4;

    v8f acc[2][4];
#pragma unroll
    for (int mi = 0; mi < 2; ++mi)
#pragma unroll
        for (int ni = 0; ni < 4; ++ni)
            acc[mi][ni] = (v8f){0.f,0.f,0.f,0.f,0.f,0.f,0.f,0.f};

    auto stage = [&](int k0, int buf) {
        // A tile: f32 -> bf16, K-contiguous rows
        {
            const int tc = (tid & 7) * 4;
            const int tr = tid >> 3;
#pragma unroll
            for (int it = 0; it < 4; ++it) {
                const int row = tr + it * 32;
                const float* src = A + (size_t)(m0 + row) * K + k0 + tc;
                f32x4 f = *reinterpret_cast<const f32x4*>(src);
                unsigned int p0 = ((unsigned)f32_to_bf16(f.y) << 16) | f32_to_bf16(f.x);
                unsigned int p1 = ((unsigned)f32_to_bf16(f.w) << 16) | f32_to_bf16(f.z);
                unsigned int* dst = reinterpret_cast<unsigned int*>(&As[buf][row][tc]);
                dst[0] = p0; dst[1] = p1;
            }
        }
        // B tile transposed to [n][k]
        {
            const int n  = tid & 127;
            const int kb = (tid >> 7) * 16;
            const int gn = n0 + n;
#pragma unroll
            for (int k = 0; k < 16; ++k) {
                const int gk = k0 + kb + k;
                float v;
                if (HEAD_DIM > 0) {
                    const int hh = gn / HD;
                    v = W[(size_t)hh * K * HD + (size_t)gk * HD + (gn % HD)];
                } else {
                    v = W[(size_t)gk * N + gn];
                }
                Bs[buf][n][kb + k] = f32_to_bf16(v);
            }
        }
    };

    const int nt = K >> 5;           // K/32 tiles
    stage(0, 0);
    for (int it = 0; it < nt; ++it) {
        const int buf = it & 1;
        __syncthreads();
        if (it + 1 < nt) stage((it + 1) * 32, buf ^ 1);  // overlaps with WMMA below

        Frag a[2], b[4];
#pragma unroll
        for (int mi = 0; mi < 2; ++mi) {
            const u32x4* ap = reinterpret_cast<const u32x4*>(&As[buf][wm + mi * 16 + l16][0]);
            a[mi].q[0] = ap[half];        // lanes<16: k0-7 ; lanes>=16: k8-15
            a[mi].q[1] = ap[half + 2];    // k16-23 / k24-31
        }
#pragma unroll
        for (int ni = 0; ni < 4; ++ni) {
            const u32x4* bp = reinterpret_cast<const u32x4*>(&Bs[buf][wn + ni * 16 + l16][0]);
            b[ni].q[0] = bp[half];        // lanes<16: k0-7 ; lanes>=16: k8-15
            b[ni].q[1] = bp[half + 2];    // k16-23 / k24-31
        }
#pragma unroll
        for (int mi = 0; mi < 2; ++mi)
#pragma unroll
            for (int ni = 0; ni < 4; ++ni)
                acc[mi][ni] = wmma_bf16(a[mi], b[ni], acc[mi][ni]);
    }

    // ---- epilogue ----
    const int rofs = half * 8;
#pragma unroll
    for (int mi = 0; mi < 2; ++mi)
#pragma unroll
        for (int ni = 0; ni < 4; ++ni) {
            const int col = n0 + wn + ni * 16 + l16;
            const float bv = bias[col];
#pragma unroll
            for (int r = 0; r < 8; ++r) {
                const int row = m0 + wm + mi * 16 + r + rofs;
                float v = acc[mi][ni][r] + bv;
                if (RELU)  v = fmaxf(v, 0.f);
                if (RESID) v += resid[(size_t)row * N + col];
                if (OUT_MODE == 1) {          // bf16 [B, H, S, hd]
                    const int bI = row / seqS, s = row % seqS;
                    const int hh = col / HD, dk = col % HD;
                    const int Hn = N / HD;
                    ((unsigned short*)outp)[((size_t)(bI * Hn + hh) * seqS + s) * HD + dk]
                        = f32_to_bf16(v);
                } else if (OUT_MODE == 2) {   // bf16 [B, H, hd, S] (transposed)
                    const int bI = row / seqS, s = row % seqS;
                    const int hh = col / HD, dk = col % HD;
                    const int Hn = N / HD;
                    ((unsigned short*)outp)[((size_t)(bI * Hn + hh) * HD + dk) * seqS + s]
                        = f32_to_bf16(v);
                } else {
                    ((float*)outp)[(size_t)row * N + col] = v;
                }
            }
        }
}

// =====================================================================
// Flash attention: one block per (128 q-rows, b*H+h). 64-key tiles,
// double-buffered async (ASYNCcnt) K/V staging.
// Qh/Kh: bf16 [B,H,S,128]; Vh: bf16 [B,H,256,S]; ctx: f32 [B,S,H*256].
// =====================================================================
__global__ __launch_bounds__(256) void attn_kernel(
    const unsigned short* __restrict__ Qh, const unsigned short* __restrict__ Kh,
    const unsigned short* __restrict__ Vh, float* __restrict__ ctx)
{
    __shared__ __align__(16) unsigned short Ks[2][64][128];   // [key][dk]  32KB
    __shared__ __align__(16) unsigned short Vt[2][256][64];   // [dv][key]  64KB
    __shared__ __align__(16) unsigned short Ps[8][16][64];    // per-wave P 16KB

    const int tid  = threadIdx.x;
    const int lane = tid & 31;
    const int wave = tid >> 5;
    const int l16  = lane & 15;
    const int half = lane >> 4;
    const int bh   = blockIdx.y;
    const int q0   = blockIdx.x * 128;
    const size_t qk_base = (size_t)bh * Sc * DKc;
    const size_t v_base  = (size_t)bh * DVc * Sc;   // transposed layout

    // Q fragments live in registers for the whole kernel (4 x 16x32 slices)
    Frag aq[4];
    {
        const int row = q0 + wave * 16 + l16;
        const unsigned short* qp = Qh + qk_base + (size_t)row * DKc;
#pragma unroll
        for (int kk = 0; kk < 4; ++kk) {
            const int off = kk * 32 + (half ? 8 : 0);
            aq[kk].q[0] = *reinterpret_cast<const u32x4*>(qp + off);
            aq[kk].q[1] = *reinterpret_cast<const u32x4*>(qp + off + 16);
        }
    }

    const unsigned long long kgbase = (unsigned long long)(Kh + qk_base);
    const unsigned long long vgbase = (unsigned long long)(Vh + v_base);

    auto stage_tiles = [&](int j, int buf) {
        // K tile 64x128: 16B per lane, 4 async b128 per thread
        const int kr = tid >> 4;            // 0..15
        const int kc = (tid & 15) * 8;      // shorts
#pragma unroll
        for (int it = 0; it < 4; ++it) {
            const int r = kr + it * 16;
            unsigned goff  = (unsigned)(((j + r) * DKc + kc) * 2);
            unsigned laddr = (unsigned)(uintptr_t)&Ks[buf][r][kc];
            async_copy_b128(laddr, goff, kgbase);
        }
        // V tile 256x64 (straight 2D copy thanks to [B,H,DV,S] layout)
        const int chunk = tid & 7;          // 8 x 16B per 128B row
        const int dvr   = tid >> 3;         // 0..31
#pragma unroll
        for (int p = 0; p < 8; ++p) {
            const int dv = dvr + p * 32;
            unsigned goff  = (unsigned)((dv * Sc + j) * 2 + chunk * 16);
            unsigned laddr = (unsigned)(uintptr_t)&Vt[buf][dv][chunk * 8];
            async_copy_b128(laddr, goff, vgbase);
        }
    };

    v8f o[16];
#pragma unroll
    for (int i = 0; i < 16; ++i) o[i] = (v8f){0.f,0.f,0.f,0.f,0.f,0.f,0.f,0.f};
    float m_run[8], l_run[8];
#pragma unroll
    for (int r = 0; r < 8; ++r) { m_run[r] = -3.0e38f; l_run[r] = 0.f; }

    stage_tiles(0, 0);
    for (int j = 0, it = 0; j < Sc; j += 64, ++it) {
        const int buf = it & 1;
        wait_async0();        // our async batch for `buf` has landed
        __syncthreads();      // everyone's batch has landed
        if (j + 64 < Sc) stage_tiles(j + 64, buf ^ 1);   // overlaps WMMA below

        // ---- scores: S = Q Kt  (16 rows x 64 keys per wave) ----
        v8f s4[4];
#pragma unroll
        for (int kt = 0; kt < 4; ++kt) s4[kt] = (v8f){0.f,0.f,0.f,0.f,0.f,0.f,0.f,0.f};
#pragma unroll
        for (int kk = 0; kk < 4; ++kk) {
            Frag bk[4];
#pragma unroll
            for (int kt = 0; kt < 4; ++kt) {
                const u32x4* bp = reinterpret_cast<const u32x4*>(
                    &Ks[buf][kt * 16 + l16][kk * 32 + half * 8]);
                bk[kt].q[0] = bp[0]; bk[kt].q[1] = bp[2];
            }
#pragma unroll
            for (int kt = 0; kt < 4; ++kt)
                s4[kt] = wmma_bf16(aq[kk], bk[kt], s4[kt]);
        }

        // ---- online softmax (vgpr r -> row r + 8*half) ----
        float corr[8];
#pragma unroll
        for (int r = 0; r < 8; ++r) {
            float mx = -3.0e38f;
#pragma unroll
            for (int kt = 0; kt < 4; ++kt) {
                s4[kt][r] = s4[kt][r] * ATTN_SCALE;
                mx = fmaxf(mx, s4[kt][r]);
            }
#pragma unroll
            for (int msk = 1; msk <= 8; msk <<= 1)
                mx = fmaxf(mx, __shfl_xor(mx, msk, 32));
            const float mn = fmaxf(m_run[r], mx);
            const float c  = __expf(m_run[r] - mn);
            m_run[r] = mn;
            corr[r]  = c;
            float ps = 0.f;
#pragma unroll
            for (int kt = 0; kt < 4; ++kt) {
                const float p = __expf(s4[kt][r] - mn);
                s4[kt][r] = p;
                ps += p;
            }
#pragma unroll
            for (int msk = 1; msk <= 8; msk <<= 1)
                ps += __shfl_xor(ps, msk, 32);
            l_run[r] = l_run[r] * c + ps;
        }
#pragma unroll
        for (int dvt = 0; dvt < 16; ++dvt)
#pragma unroll
            for (int r = 0; r < 8; ++r)
                o[dvt][r] *= corr[r];

        // ---- P (C-layout) -> LDS (A-operand source) ----
#pragma unroll
        for (int kt = 0; kt < 4; ++kt)
#pragma unroll
            for (int r = 0; r < 8; ++r)
                Ps[wave][r + half * 8][kt * 16 + l16] = f32_to_bf16(s4[kt][r]);
        __syncthreads();

        // ---- O += P @ Vtile (B-frags loaded 4 ahead of the WMMAs) ----
#pragma unroll
        for (int kk2 = 0; kk2 < 2; ++kk2) {
            Frag ap;
            const unsigned short* pp = &Ps[wave][l16][kk2 * 32 + (half ? 8 : 0)];
            ap.q[0] = *reinterpret_cast<const u32x4*>(pp);
            ap.q[1] = *reinterpret_cast<const u32x4*>(pp + 16);
#pragma unroll
            for (int g = 0; g < 4; ++g) {
                Frag bv[4];
#pragma unroll
                for (int u = 0; u < 4; ++u) {
                    const int dvt = g * 4 + u;
                    const u32x4* bp = reinterpret_cast<const u32x4*>(
                        &Vt[buf][dvt * 16 + l16][kk2 * 32 + half * 8]);
                    bv[u].q[0] = bp[0]; bv[u].q[1] = bp[2];
                }
#pragma unroll
                for (int u = 0; u < 4; ++u)
                    o[g * 4 + u] = wmma_bf16(ap, bv[u], o[g * 4 + u]);
            }
        }
    }

    // ---- finalize: ctx[b, s, h*DV + dv] = O / l ----
    const int b = bh / Hc;
    const int h = bh % Hc;
#pragma unroll
    for (int r = 0; r < 8; ++r) {
        const float inv = 1.f / l_run[r];
        const int row = q0 + wave * 16 + r + half * 8;
        const size_t base = ((size_t)b * Sc + row) * (Hc * DVc) + (size_t)h * DVc;
#pragma unroll
        for (int dvt = 0; dvt < 16; ++dvt)
            ctx[base + dvt * 16 + l16] = o[dvt][r] * inv;
    }
}

// =====================================================================
// Row-wise LayerNorm: one block (256 thr) per row of length Dim.
// =====================================================================
__global__ __launch_bounds__(256) void ln_kernel(
    const float* __restrict__ x, const float* __restrict__ g,
    const float* __restrict__ bt, float* __restrict__ out, int Dim)
{
    __shared__ float red[256];
    const int row = blockIdx.x;
    const int tid = threadIdx.x;
    const float* xr = x + (size_t)row * Dim;

    float s = 0.f;
    for (int i = tid; i < Dim; i += 256) s += xr[i];
    red[tid] = s; __syncthreads();
    for (int off = 128; off > 0; off >>= 1) {
        if (tid < off) red[tid] += red[tid + off];
        __syncthreads();
    }
    const float mu = red[0] / Dim;
    __syncthreads();

    float v = 0.f;
    for (int i = tid; i < Dim; i += 256) { const float d = xr[i] - mu; v += d * d; }
    red[tid] = v; __syncthreads();
    for (int off = 128; off > 0; off >>= 1) {
        if (tid < off) red[tid] += red[tid + off];
        __syncthreads();
    }
    const float rstd = rsqrtf(red[0] / Dim + 1e-5f);
    for (int i = tid; i < Dim; i += 256)
        out[(size_t)row * Dim + i] = (xr[i] - mu) * rstd * g[i] + bt[i];
}

// =====================================================================
extern "C" void kernel_launch(void* const* d_in, const int* in_sizes, int n_in,
                              void* d_out, int out_size, void* d_ws, size_t ws_size,
                              hipStream_t stream) {
    const float* x   = (const float*)d_in[0];
    const float* Wq  = (const float*)d_in[1];
    const float* bq  = (const float*)d_in[2];
    const float* Wk  = (const float*)d_in[3];
    const float* bk  = (const float*)d_in[4];
    const float* Wv  = (const float*)d_in[5];
    const float* bv  = (const float*)d_in[6];
    const float* WQp = (const float*)d_in[7];
    const float* bQp = (const float*)d_in[8];
    const float* WKp = (const float*)d_in[9];
    const float* bKp = (const float*)d_in[10];
    const float* WVp = (const float*)d_in[11];
    const float* bVp = (const float*)d_in[12];
    const float* Wo  = (const float*)d_in[13];
    const float* bo  = (const float*)d_in[14];
    const float* W1  = (const float*)d_in[15];
    const float* b1  = (const float*)d_in[16];
    const float* W2  = (const float*)d_in[17];
    const float* b2  = (const float*)d_in[18];
    const float* g1  = (const float*)d_in[19];
    const float* be1 = (const float*)d_in[20];
    const float* g2  = (const float*)d_in[21];
    const float* be2 = (const float*)d_in[22];

    // ---- workspace carve (≈144 MB) ----
    char* w = (char*)d_ws;
    const size_t MB = 1024ull * 1024ull;
    float*          Q    = (float*)(w + 0 * MB);     // [4096,512]  f32  8MB
    float*          Kf   = (float*)(w + 8 * MB);     // [4096,512]  f32  8MB
    float*          Vf   = (float*)(w + 16 * MB);    // [4096,512]  f32  8MB
    unsigned short* Qh   = (unsigned short*)(w + 24 * MB);  // bf16 [B,H,S,128]  8MB
    unsigned short* Kh   = (unsigned short*)(w + 32 * MB);  // bf16 [B,H,S,128]  8MB
    unsigned short* Vh   = (unsigned short*)(w + 40 * MB);  // bf16 [B,H,256,S] 16MB
    float*          ctx  = (float*)(w + 56 * MB);    // [4096,2048] f32 32MB
    float*          y1   = (float*)(w + 88 * MB);    // x+attn_out       8MB
    float*          hbuf = (float*)(w + 96 * MB);    // LN1 out          8MB
    float*          ff1  = (float*)(w + 104 * MB);   // relu(h W1+b1)   32MB
    float*          y2   = (float*)(w + 136 * MB);   // h+ff             8MB

    const dim3 blk(256);

    // 1) QKV projections (f32 out)
    gemm_bf16_kernel<0, 0, false, false><<<dim3(Dc / 128, Mc / 128), blk, 0, stream>>>(
        x, Wq, bq, nullptr, Q, Mc, Dc, Dc, Sc);
    gemm_bf16_kernel<0, 0, false, false><<<dim3(Dc / 128, Mc / 128), blk, 0, stream>>>(
        x, Wk, bk, nullptr, Kf, Mc, Dc, Dc, Sc);
    gemm_bf16_kernel<0, 0, false, false><<<dim3(Dc / 128, Mc / 128), blk, 0, stream>>>(
        x, Wv, bv, nullptr, Vf, Mc, Dc, Dc, Sc);

    // 2) per-head projections -> bf16 (Q/K row-major, V transposed per head)
    gemm_bf16_kernel<128, 1, false, false><<<dim3((Hc * DKc) / 128, Mc / 128), blk, 0, stream>>>(
        Q, WQp, bQp, nullptr, Qh, Mc, Hc * DKc, Dc, Sc);
    gemm_bf16_kernel<128, 1, false, false><<<dim3((Hc * DKc) / 128, Mc / 128), blk, 0, stream>>>(
        Kf, WKp, bKp, nullptr, Kh, Mc, Hc * DKc, Dc, Sc);
    gemm_bf16_kernel<256, 2, false, false><<<dim3((Hc * DVc) / 128, Mc / 128), blk, 0, stream>>>(
        Vf, WVp, bVp, nullptr, Vh, Mc, Hc * DVc, Dc, Sc);

    // 3) flash attention -> ctx [B,S,H*DV] f32
    attn_kernel<<<dim3(Sc / 128, Bc * Hc), blk, 0, stream>>>(Qh, Kh, Vh, ctx);

    // 4) output projection + residual(x)
    gemm_bf16_kernel<0, 0, false, true><<<dim3(Dc / 128, Mc / 128), blk, 0, stream>>>(
        ctx, Wo, bo, x, y1, Mc, Dc, Hc * DVc, Sc);

    // 5) LN1
    ln_kernel<<<Mc, blk, 0, stream>>>(y1, g1, be1, hbuf, Dc);

    // 6) FFN
    gemm_bf16_kernel<0, 0, true, false><<<dim3(DFFc / 128, Mc / 128), blk, 0, stream>>>(
        hbuf, W1, b1, nullptr, ff1, Mc, DFFc, Dc, Sc);
    gemm_bf16_kernel<0, 0, false, true><<<dim3(Dc / 128, Mc / 128), blk, 0, stream>>>(
        ff1, W2, b2, hbuf, y2, Mc, Dc, DFFc, Sc);

    // 7) LN2 -> output
    ln_kernel<<<Mc, blk, 0, stream>>>(y2, g2, be2, (float*)d_out, Dc);
}
